// QuantLinear_81449759801528
// MI455X (gfx1250) — compile-verified, matrix operand order checked by reading, not run on hardware
//
#include <hip/hip_runtime.h>

// Problem constants (match reference)
#define OUT_F 4096
#define IN_F  4096
#define MTOT  8192          // 4 * 2048
#define NGRP  32            // IN_F / 128

// GEMM tiling: block 128(M) x 256(N), 8 wave32 in a 2x4 grid, wave tile 64x64
#define BM 128
#define BN 256
#define BK 32
#define BKP 40              // padded LDS stride (80B) -> conflict-free, 16B aligned
#define KTILES (IN_F / BK)  // 128

typedef __attribute__((ext_vector_type(16))) _Float16     v16h;
typedef __attribute__((ext_vector_type(8)))  _Float16     v8h;
typedef __attribute__((ext_vector_type(8)))  float        v8f;
typedef __attribute__((ext_vector_type(4)))  unsigned int v4u;
typedef __attribute__((ext_vector_type(4)))  int          v4i;

// Address-space helper types for the async-to-LDS builtin
typedef __attribute__((address_space(1))) void GVoid;   // global
typedef __attribute__((address_space(3))) void LVoid;   // LDS
typedef __attribute__((address_space(1))) v4i  GVec4;   // global int4*
typedef __attribute__((address_space(3))) v4i  LVec4;   // LDS int4*

#if __has_builtin(__builtin_amdgcn_global_load_async_to_lds_b128) && \
    __has_builtin(__builtin_amdgcn_s_wait_asynccnt)
#define USE_ASYNC 1
#else
#define USE_ASYNC 0
#endif

union Frag {
    v16h h;
    v4u  u[2];
};

// ---------------------------------------------------------------------------
// Phase 1a: dequantize weights to f16, layout [OUT_F][IN_F] (row-major, K fast)
// w[o][i] = code[o][i] * scale[o][i/128] - offset[o][i/128]
// ---------------------------------------------------------------------------
__global__ __launch_bounds__(256) void dequant_w_f16(
    const int* __restrict__ cw, const float* __restrict__ scale,
    const float* __restrict__ offs, _Float16* __restrict__ Wh) {
    size_t base = ((size_t)blockIdx.x * 256 + threadIdx.x) * 8;
    int o = (int)(base >> 12);          // / IN_F
    int i = (int)(base & (IN_F - 1));
    int g = i >> 7;                     // / 128
    float s = scale[(size_t)o * NGRP + g];
    float f = offs[(size_t)o * NGRP + g];
    const int4* cp = (const int4*)(cw + base);
    int4 c0 = cp[0];
    int4 c1 = cp[1];
    v8h h;
    h[0] = (_Float16)((float)c0.x * s - f);
    h[1] = (_Float16)((float)c0.y * s - f);
    h[2] = (_Float16)((float)c0.z * s - f);
    h[3] = (_Float16)((float)c0.w * s - f);
    h[4] = (_Float16)((float)c1.x * s - f);
    h[5] = (_Float16)((float)c1.y * s - f);
    h[6] = (_Float16)((float)c1.z * s - f);
    h[7] = (_Float16)((float)c1.w * s - f);
    *(v8h*)(Wh + base) = h;
}

// ---------------------------------------------------------------------------
// Phase 1b: x f32 -> f16 (so the GEMM working set is 96MB and fits in 192MB L2)
// ---------------------------------------------------------------------------
__global__ __launch_bounds__(256) void cvt_x_f16(
    const float* __restrict__ X, _Float16* __restrict__ Xh) {
    size_t base = ((size_t)blockIdx.x * 256 + threadIdx.x) * 8;
    const float4* xp = (const float4*)(X + base);
    float4 a = xp[0];
    float4 b = xp[1];
    v8h h;
    h[0] = (_Float16)a.x; h[1] = (_Float16)a.y;
    h[2] = (_Float16)a.z; h[3] = (_Float16)a.w;
    h[4] = (_Float16)b.x; h[5] = (_Float16)b.y;
    h[6] = (_Float16)b.z; h[7] = (_Float16)b.w;
    *(v8h*)(Xh + base) = h;
}

// ---------------------------------------------------------------------------
// Phase 2: f16 WMMA GEMM.  out[M,N] = Xh[M,K] * Wh[N,K]^T
// Block tile 128x256, 256 threads = 8 wave32 arranged 2(M) x 4(N); each wave
// owns a 64x64 tile = 4x4 v_wmma_f32_16x16x32_f16. A-fragments are held for
// the whole k-step; B-fragments are streamed (one live at a time) to keep the
// live set < 256 VGPRs and the accumulator bank pinned (no rotation movs).
// Double-buffered LDS, BK=32; fill via GLOBAL_LOAD_ASYNC_TO_LDS_B128.
// ---------------------------------------------------------------------------
__global__ __launch_bounds__(256) void gemm_q4_f16(
    const _Float16* __restrict__ Xh, const _Float16* __restrict__ Wh,
    float* __restrict__ out) {
    __shared__ __align__(16) _Float16 As[2][BM][BKP];
    __shared__ __align__(16) _Float16 Bs[2][BN][BKP];

    const int tid   = threadIdx.x;
    const int lane  = tid & 31;
    const int wave  = tid >> 5;
    const int waveM = wave >> 2;   // 0..1
    const int waveN = wave & 3;    // 0..3
    const int m0    = blockIdx.y * BM;
    const int n0    = blockIdx.x * BN;

    // cooperative copy mapping: 16B chunks; 4 chunks per 32-half row
    const int chunkh = (tid & 3) * 8;   // half offset within row: 0,8,16,24
    const int rowc   = tid >> 2;        // 0..63

    v8f acc[4][4];
#pragma unroll
    for (int m = 0; m < 4; ++m)
#pragma unroll
        for (int n = 0; n < 4; ++n)
#pragma unroll
            for (int e = 0; e < 8; ++e) acc[m][n][e] = 0.0f;

#if USE_ASYNC
    // A: 2 rows/thread (128 rows), B: 4 rows/thread (256 rows); all 16B async
    // copies straight to LDS -- no VGPR staging, tracked on ASYNCcnt.
    auto issueStage = [&](int kt, int buf) {
        const size_t k0 = (size_t)kt * BK;
#pragma unroll
        for (int j = 0; j < 2; ++j) {
            const _Float16* ga = Xh + (size_t)(m0 + rowc + 64 * j) * IN_F + k0 + chunkh;
            __builtin_amdgcn_global_load_async_to_lds_b128(
                (GVec4*)(GVoid*)ga,
                (LVec4*)(LVoid*)&As[buf][rowc + 64 * j][chunkh], 0, 0);
        }
#pragma unroll
        for (int j = 0; j < 4; ++j) {
            const _Float16* gb = Wh + (size_t)(n0 + rowc + 64 * j) * IN_F + k0 + chunkh;
            __builtin_amdgcn_global_load_async_to_lds_b128(
                (GVec4*)(GVoid*)gb,
                (LVec4*)(LVoid*)&Bs[buf][rowc + 64 * j][chunkh], 0, 0);
        }
    };
#else
    v4u ra[2], rb[4];
    auto loadG = [&](int kt) {
        const size_t k0 = (size_t)kt * BK;
#pragma unroll
        for (int j = 0; j < 2; ++j)
            ra[j] = *(const v4u*)(Xh + (size_t)(m0 + rowc + 64 * j) * IN_F + k0 + chunkh);
#pragma unroll
        for (int j = 0; j < 4; ++j)
            rb[j] = *(const v4u*)(Wh + (size_t)(n0 + rowc + 64 * j) * IN_F + k0 + chunkh);
    };
    auto storeL = [&](int buf) {
#pragma unroll
        for (int j = 0; j < 2; ++j)
            *(v4u*)&As[buf][rowc + 64 * j][chunkh] = ra[j];
#pragma unroll
        for (int j = 0; j < 4; ++j)
            *(v4u*)&Bs[buf][rowc + 64 * j][chunkh] = rb[j];
    };
#endif

    auto compute = [&](int buf) {
        // ISA 16-bit A-matrix 16x32 layout: lanes 0-15 take K {0..7,16..23},
        // lanes 16-31 take K {8..15,24..31}; two 16B LDS loads per fragment.
        const int kb = (lane < 16) ? 0 : 8;
        const int r  = lane & 15;
        Frag a[4];
#pragma unroll
        for (int m = 0; m < 4; ++m) {
            const _Float16* p = &As[buf][waveM * 64 + m * 16 + r][kb];
            a[m].u[0] = *(const v4u*)p;
            a[m].u[1] = *(const v4u*)(p + 16);
        }
#pragma unroll
        for (int n = 0; n < 4; ++n) {
            Frag b;
            const _Float16* p = &Bs[buf][waveN * 64 + n * 16 + r][kb];
            b.u[0] = *(const v4u*)p;
            b.u[1] = *(const v4u*)(p + 16);
#pragma unroll
            for (int m = 0; m < 4; ++m)
                acc[m][n] = __builtin_amdgcn_wmma_f32_16x16x32_f16(
                    false, a[m].h, false, b.h,
                    (short)0, acc[m][n], false, false);
        }
    };

    // software pipeline: async fill of stage kt+1 (buffer buf^1) overlaps compute(kt).
    // unroll 1: keep the accumulator bank pinned across iterations (no rotation).
#if USE_ASYNC
    issueStage(0, 0);
    __builtin_amdgcn_s_wait_asynccnt(0);
    __syncthreads();
#else
    loadG(0);
    storeL(0);
    __syncthreads();
#endif
    int buf = 0;
#pragma unroll 1
    for (int kt = 0; kt < KTILES; ++kt) {
#if USE_ASYNC
        if (kt + 1 < KTILES) issueStage(kt + 1, buf ^ 1);
        compute(buf);
        if (kt + 1 < KTILES) {
            __builtin_amdgcn_s_wait_asynccnt(0);
            __syncthreads();
        }
#else
        if (kt + 1 < KTILES) loadG(kt + 1);
        compute(buf);
        if (kt + 1 < KTILES) {
            storeL(buf ^ 1);
            __syncthreads();
        }
#endif
        buf ^= 1;
    }

    // epilogue: C layout (VGPR e: lanes 0-15 -> M=e, lanes 16-31 -> M=8+e; N=lane&15)
    const int mhi = (lane >> 4) << 3;
#pragma unroll
    for (int m = 0; m < 4; ++m)
#pragma unroll
        for (int n = 0; n < 4; ++n) {
            const int rbase = m0 + waveM * 64 + m * 16 + mhi;
            const int cbase = n0 + waveN * 64 + n * 16 + (lane & 15);
#pragma unroll
            for (int e = 0; e < 8; ++e)
                out[(size_t)(rbase + e) * OUT_F + cbase] = acc[m][n][e];
        }
}

// ---------------------------------------------------------------------------
extern "C" void kernel_launch(void* const* d_in, const int* in_sizes, int n_in,
                              void* d_out, int out_size, void* d_ws, size_t ws_size,
                              hipStream_t stream) {
    (void)in_sizes; (void)n_in; (void)out_size; (void)ws_size;
    const float* x     = (const float*)d_in[0];
    const int*   cw    = (const int*)d_in[1];
    const float* scale = (const float*)d_in[2];
    const float* offs  = (const float*)d_in[3];
    float*       out   = (float*)d_out;

    // workspace layout: [0, 32MB) Wh f16 [OUT_F][IN_F]; [32MB, 96MB) Xh f16 [MTOT][IN_F]
    _Float16* Wh = (_Float16*)d_ws;
    _Float16* Xh = (_Float16*)((char*)d_ws + (size_t)OUT_F * IN_F * sizeof(_Float16));

    dequant_w_f16<<<(OUT_F * (size_t)IN_F) / 8 / 256, 256, 0, stream>>>(cw, scale, offs, Wh);
    cvt_x_f16<<<((size_t)MTOT * IN_F) / 8 / 256, 256, 0, stream>>>(x, Xh);

    dim3 grid(OUT_F / BN, MTOT / BM);
    gemm_q4_f16<<<grid, 256, 0, stream>>>(Xh, Wh, out);
}
